// Generator_57509612093612
// MI455X (gfx1250) — compile-verified, matrix operand order checked by reading, not run on hardware
//
#include <hip/hip_runtime.h>
#include <hip/hip_bf16.h>
#include <stdint.h>

// ---------------------------------------------------------------------------
// MI455X (gfx1250) implementation.
// BL = 256 batch elements, each a 256x256 matrix. All six 256^3 GEMMs run on
// v_wmma_f32_16x16x32_bf16. Each wave owns a 16x64 tile (4 f32 accumulators),
// reusing one A-fragment across 4 B-fragments (4 WMMA per A ds_load).
// A-panels (32x256 bf16, 16 KB) are filled with GLOBAL_LOAD_ASYNC_TO_LDS_B128
// when the toolchain exposes it (ASYNCcnt path, no VGPR staging).
// ---------------------------------------------------------------------------

typedef __attribute__((ext_vector_type(16))) __bf16 v16bf;
typedef __attribute__((ext_vector_type(8)))  float  v8f;

#define N 256
#define NN 65536           // 256*256
#define BL 256             // B*L
#define TOPK 32768         // 0.5 * N * N

#if __has_builtin(__builtin_amdgcn_global_load_async_to_lds_b128)
#define HAVE_ASYNC_LDS 1
// Matches the builtin's parameter type as printed by clang:
//   '__attribute__((__vector_size__(4 * sizeof(int)))) int *'
typedef int v4i_vec __attribute__((vector_size(4 * sizeof(int))));
#endif

__device__ __forceinline__ unsigned short f32_to_bf16(float f) {
    unsigned u = __float_as_uint(f);
    u += 0x7FFFu + ((u >> 16) & 1u);        // round to nearest even
    return (unsigned short)(u >> 16);
}

__device__ __forceinline__ unsigned f32_to_key(float f) {
    unsigned u = __float_as_uint(f);
    return (u & 0x80000000u) ? ~u : (u | 0x80000000u);   // order-preserving
}

__device__ __forceinline__ float wave_max(float v) {
    #pragma unroll
    for (int m = 16; m > 0; m >>= 1) v = fmaxf(v, __shfl_xor(v, m, 32));
    return v;
}
__device__ __forceinline__ float wave_sum(float v) {
    #pragma unroll
    for (int m = 16; m > 0; m >>= 1) v += __shfl_xor(v, m, 32);
    return v;
}

// Cooperative fill of a 32-row x 256-col bf16 A-panel (16 KB) in LDS.
__device__ __forceinline__ void load_panel32(unsigned short* sA,
                                             const unsigned short* gA) {
#ifdef HAVE_ASYNC_LDS
    for (int i = threadIdx.x; i < 1024; i += 256)   // 1024 x 16B chunks
        __builtin_amdgcn_global_load_async_to_lds_b128(
            (v4i_vec*)(gA + i * 8), (v4i_vec*)(sA + i * 8), 0, 0);
#if __has_builtin(__builtin_amdgcn_s_wait_asynccnt)
    __builtin_amdgcn_s_wait_asynccnt(0);
#else
    asm volatile("s_wait_asynccnt 0" ::: "memory");
#endif
    __syncthreads();
#else
    const uint4* src = (const uint4*)gA;            // 1024 x uint4
    uint4* dst = (uint4*)sA;
    for (int i = threadIdx.x; i < 1024; i += 256) dst[i] = src[i];
    __syncthreads();
#endif
}

// One wave computes a 16x64 f32 tile (4 accumulators), sharing the A-fragment.
// A-frag: lane L holds row (rbase + (L&15)), 16 contiguous k at ((L>>4)<<4).
// B-frag: lane L holds col, 16 contiguous k -> B stored column-major
//         as Bcm[col*256 + k].
__device__ __forceinline__ void wave_gemm4(const unsigned short* sA, int rbase,
                                           const unsigned short* Bcm, int col0,
                                           int lane, v8f c[4]) {
    const int khalf = (lane >> 4) << 4;
    const unsigned short* ap = sA + (rbase + (lane & 15)) * N + khalf;
    const unsigned short* bp = Bcm + (col0 + (lane & 15)) * N + khalf;
    #pragma unroll
    for (int kb = 0; kb < N; kb += 32) {
        v16bf a = *(const v16bf*)(ap + kb);               // ds_load (LDS)
        #pragma unroll
        for (int t = 0; t < 4; ++t) {
            __builtin_prefetch(bp + t * 16 * N + kb + 64, 0, 0);
            v16bf b = *(const v16bf*)(bp + t * 16 * N + kb);  // global_load
            c[t] = __builtin_amdgcn_wmma_f32_16x16x32_bf16(
                       false, a, false, b, (short)0, c[t], false, false);
        }
    }
}

// Wave/tile geometry shared by all GEMM kernels:
//   block = 256 threads = 8 waves in a 2(row) x 4(col) arrangement
//   block tile = 32 rows x 256 cols; gridDim.y = 8 row panels
struct TileGeom { int rbase, row0, col0, lane, n, mbase; };
__device__ __forceinline__ TileGeom tile_geom() {
    TileGeom g;
    const int wave = threadIdx.x >> 5;
    g.lane  = threadIdx.x & 31;
    g.rbase = (wave >> 2) * 16;                 // row offset inside panel
    g.row0  = blockIdx.y * 32 + g.rbase;        // global row base of tile
    g.col0  = (wave & 3) * 64;                  // global col base of tile
    g.n     = g.lane & 15;
    g.mbase = (g.lane >> 4) << 3;
    return g;
}

// --------------------------- conversion kernels ----------------------------

__global__ void k_convert_x(const float* __restrict__ feature,
                            unsigned short* __restrict__ xb,
                            unsigned short* __restrict__ xt) {
    int idx = blockIdx.x * 256 + threadIdx.x;      // over BL*N*N
    float f = feature[idx];
    unsigned short v = f32_to_bf16(f);
    xb[idx] = v;
    int b = idx >> 16, rc = idx & (NN - 1);
    int r = rc >> 8, col = rc & 255;
    xt[(b << 16) + (col << 8) + r] = v;            // x^T, row-major in col
}

__global__ void k_convert_w(const float* __restrict__ Wq,
                            const float* __restrict__ Wk,
                            const float* __restrict__ Wv,
                            unsigned short* __restrict__ wbf) {
    int idx = blockIdx.x * 256 + threadIdx.x;      // over 3*N*N
    int which = idx >> 16, e = idx & (NN - 1);
    const float* src = which == 0 ? Wq : (which == 1 ? Wk : Wv);
    wbf[idx] = f32_to_bf16(src[e]);
}

__global__ void k_rowcolsq(const float* __restrict__ feature,
                           float* __restrict__ rowsq,
                           float* __restrict__ colsq) {
    int idx = blockIdx.x * 256 + threadIdx.x;      // over 2*BL*N
    int which = idx >> 16;                          // 0 = row, 1 = col
    int e = idx & (NN - 1);
    int b = e >> 8, i = e & 255;
    const float* base = feature + (size_t)b * NN;
    float s = 0.f;
    if (which == 0) {
        #pragma unroll 4
        for (int k = 0; k < N; ++k) { float v = base[i * N + k]; s += v * v; }
        rowsq[e] = s;
    } else {
        #pragma unroll 4
        for (int k = 0; k < N; ++k) { float v = base[k * N + i]; s += v * v; }
        colsq[e] = s;
    }
}

// ------------------------------ GEMM kernels -------------------------------

// Q = x Wq^T + bq (stored transposed bf16), K likewise, V row-major bf16.
// B operand: row n of W == column n of W^T -> W row-major is col-major B.
__global__ void k_gemm_qkv(const unsigned short* __restrict__ xb,
                           const unsigned short* __restrict__ wbf,
                           const float* __restrict__ bq,
                           const float* __restrict__ bk,
                           const float* __restrict__ bv,
                           unsigned short* __restrict__ Qt,
                           unsigned short* __restrict__ Kt,
                           unsigned short* __restrict__ Vb) {
    __shared__ __align__(16) unsigned short sA[32 * N];
    const int b = blockIdx.z / 3, which = blockIdx.z % 3;
    TileGeom g = tile_geom();
    load_panel32(sA, xb + (size_t)b * NN + blockIdx.y * 32 * N);
    v8f c[4] = {};
    wave_gemm4(sA, g.rbase, wbf + which * NN, g.col0, g.lane, c);
    const float* bias = which == 0 ? bq : (which == 1 ? bk : bv);
    if (which == 2) {
        unsigned short* out = Vb + (size_t)b * NN;
        #pragma unroll
        for (int t = 0; t < 4; ++t) {
            const int cb = g.col0 + 16 * t;
            const float bb = bias[cb + g.n];
            #pragma unroll
            for (int r = 0; r < 8; ++r)
                out[(g.row0 + g.mbase + r) * N + cb + g.n] =
                    f32_to_bf16(c[t][r] + bb);
        }
    } else {
        unsigned short* out = (which == 0 ? Qt : Kt) + (size_t)b * NN;
        #pragma unroll
        for (int t = 0; t < 4; ++t) {
            const int cb = g.col0 + 16 * t;
            const float bb = bias[cb + g.n];
            #pragma unroll
            for (int r = 0; r < 8; ++r)
                out[(cb + g.n) * N + g.row0 + g.mbase + r] =
                    f32_to_bf16(c[t][r] + bb);
        }
    }
}

// att_pre[w,v] = sum_h K[h,w] Q[h,v] / 16 :  A = K^T (Kt row-major),
// B col-major = Q^T rows (Qt row-major).
__global__ void k_gemm_att(const unsigned short* __restrict__ Kt,
                           const unsigned short* __restrict__ Qt,
                           float* __restrict__ attp) {
    __shared__ __align__(16) unsigned short sA[32 * N];
    const int b = blockIdx.z;
    TileGeom g = tile_geom();
    load_panel32(sA, Kt + (size_t)b * NN + blockIdx.y * 32 * N);
    v8f c[4] = {};
    wave_gemm4(sA, g.rbase, Qt + (size_t)b * NN, g.col0, g.lane, c);
    float* out = attp + (size_t)b * NN;
    #pragma unroll
    for (int t = 0; t < 4; ++t)
        #pragma unroll
        for (int r = 0; r < 8; ++r)
            out[(g.row0 + g.mbase + r) * N + g.col0 + 16 * t + g.n] =
                c[t][r] * 0.0625f;                         // 1/sqrt(256)
}

// Row softmax of att_pre; store transposed bf16 so the V@att GEMM reads a
// contiguous column-major B operand.
__global__ void k_softmax_att(const float* __restrict__ attp,
                              unsigned short* __restrict__ attT) {
    const int b = blockIdx.y;
    const int wave = threadIdx.x >> 5, lane = threadIdx.x & 31;
    const int row = blockIdx.x * 8 + wave;
    const float* src = attp + (size_t)b * NN + row * N;
    float v[8]; float mx = -3.0e38f;
    #pragma unroll
    for (int j = 0; j < 8; ++j) { v[j] = src[lane + 32 * j]; mx = fmaxf(mx, v[j]); }
    mx = wave_max(mx);
    float s = 0.f;
    #pragma unroll
    for (int j = 0; j < 8; ++j) { v[j] = __expf(v[j] - mx); s += v[j]; }
    s = wave_sum(s);
    const float inv = 1.f / s;
    unsigned short* dst = attT + (size_t)b * NN;
    #pragma unroll
    for (int j = 0; j < 8; ++j) {
        int col = lane + 32 * j;
        dst[col * N + row] = f32_to_bf16(v[j] * inv);
    }
}

// adj_att[h,v] = sum_w V[h,w] att[w,v] : A = V row-major, B col-major = attT.
__global__ void k_gemm_vatt(const unsigned short* __restrict__ Vb,
                            const unsigned short* __restrict__ attT,
                            float* __restrict__ adj_att) {
    __shared__ __align__(16) unsigned short sA[32 * N];
    const int b = blockIdx.z;
    TileGeom g = tile_geom();
    load_panel32(sA, Vb + (size_t)b * NN + blockIdx.y * 32 * N);
    v8f c[4] = {};
    wave_gemm4(sA, g.rbase, attT + (size_t)b * NN, g.col0, g.lane, c);
    float* out = adj_att + (size_t)b * NN;
    #pragma unroll
    for (int t = 0; t < 4; ++t)
        #pragma unroll
        for (int r = 0; r < 8; ++r)
            out[(g.row0 + g.mbase + r) * N + g.col0 + 16 * t + g.n] = c[t][r];
}

// cross = x @ x ; fuse RBF: gauss = exp(-(||r_i||^2 + ||c_j||^2 - 2 cross)/2)
__global__ void k_gemm_cross(const unsigned short* __restrict__ xb,
                             const unsigned short* __restrict__ xt,
                             const float* __restrict__ rowsq,
                             const float* __restrict__ colsq,
                             float* __restrict__ gauss) {
    __shared__ __align__(16) unsigned short sA[32 * N];
    const int b = blockIdx.z;
    TileGeom g = tile_geom();
    load_panel32(sA, xb + (size_t)b * NN + blockIdx.y * 32 * N);
    v8f c[4] = {};
    wave_gemm4(sA, g.rbase, xt + (size_t)b * NN, g.col0, g.lane, c);
    float* out = gauss + (size_t)b * NN;
    #pragma unroll
    for (int t = 0; t < 4; ++t)
        #pragma unroll
        for (int r = 0; r < 8; ++r) {
            int m = g.row0 + g.mbase + r, nn = g.col0 + 16 * t + g.n;
            float d = rowsq[b * N + m] + colsq[b * N + nn] - 2.f * c[t][r];
            out[m * N + nn] = __expf(-0.5f * d);           // SIGMA = 1
        }
}

// ------------------------- fused softmax + top-k ---------------------------

__global__ void k_final(const float* __restrict__ adj_att,
                        const float* __restrict__ gauss,
                        const float* __restrict__ gumbel,
                        unsigned* __restrict__ keys,
                        float* __restrict__ out) {
    const int b = blockIdx.x;
    const int wave = threadIdx.x >> 5, lane = threadIdx.x & 31;
    const size_t base = (size_t)b * NN;
    // Phase 1: adj = rowsoftmax(adj_att) + rowsoftmax(gauss) + gumbel -> keys
    for (int rr = 0; rr < 32; ++rr) {
        const int row = wave * 32 + rr;
        const float* pa = adj_att + base + row * N;
        const float* pg = gauss   + base + row * N;
        float av[8], gv[8], am = -3.0e38f, gm = -3.0e38f;
        #pragma unroll
        for (int j = 0; j < 8; ++j) {
            av[j] = pa[lane + 32 * j]; am = fmaxf(am, av[j]);
            gv[j] = pg[lane + 32 * j]; gm = fmaxf(gm, gv[j]);
        }
        am = wave_max(am); gm = wave_max(gm);
        float as = 0.f, gs = 0.f;
        #pragma unroll
        for (int j = 0; j < 8; ++j) {
            av[j] = __expf(av[j] - am); as += av[j];
            gv[j] = __expf(gv[j] - gm); gs += gv[j];
        }
        as = wave_sum(as); gs = wave_sum(gs);
        const float ia = 1.f / as, ig = 1.f / gs;
        #pragma unroll
        for (int j = 0; j < 8; ++j) {
            int col = lane + 32 * j;
            float val = av[j] * ia + gv[j] * ig + gumbel[base + row * N + col];
            keys[base + row * N + col] = f32_to_key(val);
        }
    }
    __syncthreads();
    // Phase 2: binary search the k-th largest key (count >= mid is monotone)
    __shared__ unsigned scnt, slo, shi;
    if (threadIdx.x == 0) { slo = 0u; shi = 0xFFFFFFFFu; }
    __syncthreads();
    for (int it = 0; it < 32; ++it) {
        unsigned lo = slo, hi = shi;
        if (lo >= hi) break;                       // uniform across block
        unsigned mid = lo + ((hi - lo) >> 1) + 1u;
        if (threadIdx.x == 0) scnt = 0u;
        __syncthreads();
        unsigned local = 0;
        for (int i = threadIdx.x; i < NN; i += 256)
            local += (keys[base + i] >= mid) ? 1u : 0u;
        atomicAdd(&scnt, local);
        __syncthreads();
        if (threadIdx.x == 0) {
            if (scnt >= (unsigned)TOPK) slo = mid; else shi = mid - 1u;
        }
        __syncthreads();
    }
    const unsigned T = slo;
    // Phase 3: mask
    for (int i = threadIdx.x; i < NN; i += 256)
        out[base + i] = (keys[base + i] >= T) ? 1.0f : 0.0f;
}

// ------------------------------- launcher ----------------------------------

extern "C" void kernel_launch(void* const* d_in, const int* in_sizes, int n_in,
                              void* d_out, int out_size, void* d_ws, size_t ws_size,
                              hipStream_t stream) {
    (void)in_sizes; (void)n_in; (void)out_size; (void)ws_size;
    const float* feature = (const float*)d_in[0];
    const float* Wq = (const float*)d_in[1];
    const float* bq = (const float*)d_in[2];
    const float* Wk = (const float*)d_in[3];
    const float* bk = (const float*)d_in[4];
    const float* Wv = (const float*)d_in[5];
    const float* bv = (const float*)d_in[6];
    const float* gumbel = (const float*)d_in[7];
    float* out = (float*)d_out;

    // workspace carve-up (all offsets 256B-aligned)
    const size_t S_BF = (size_t)BL * NN * sizeof(unsigned short); // 32 MiB
    const size_t S_F32 = (size_t)BL * NN * sizeof(float);         // 64 MiB
    uint8_t* w = (uint8_t*)d_ws;
    size_t off = 0;
    unsigned short* xb   = (unsigned short*)(w + off); off += S_BF;
    unsigned short* xt   = (unsigned short*)(w + off); off += S_BF;
    unsigned short* wbf  = (unsigned short*)(w + off); off += 3 * (size_t)NN * 2 + 256;
    float* rowsq         = (float*)(w + off); off += (size_t)BL * N * 4;
    float* colsq         = (float*)(w + off); off += (size_t)BL * N * 4;
    unsigned short* Qt   = (unsigned short*)(w + off); off += S_BF;
    unsigned short* Kt   = (unsigned short*)(w + off); off += S_BF;
    unsigned short* Vb   = (unsigned short*)(w + off); off += S_BF;
    unsigned short* attT = (unsigned short*)(w + off); off += S_BF;
    float* bufA          = (float*)(w + off); off += S_F32;  // att_pre -> gauss
    float* bufB          = (float*)(w + off); off += S_F32;  // adj_att
    // keys alias Qt+Kt (64 MiB, dead after k_gemm_att)
    unsigned* keys = (unsigned*)Qt;

    // 1) precision conversion + squared norms
    k_convert_x<<<dim3(BL * NN / 256), dim3(256), 0, stream>>>(feature, xb, xt);
    k_convert_w<<<dim3(3 * NN / 256), dim3(256), 0, stream>>>(Wq, Wk, Wv, wbf);
    k_rowcolsq<<<dim3(2 * BL * N / 256), dim3(256), 0, stream>>>(feature, rowsq, colsq);

    // 2) Q/K/V (WMMA), Q and K stored transposed for the attention GEMM
    k_gemm_qkv<<<dim3(1, 8, BL * 3), dim3(256), 0, stream>>>(
        xb, wbf, bq, bk, bv, Qt, Kt, Vb);

    // 3) att_pre = K^T Q / 16 (WMMA)
    k_gemm_att<<<dim3(1, 8, BL), dim3(256), 0, stream>>>(Kt, Qt, bufA);

    // 4) row softmax -> att^T bf16
    k_softmax_att<<<dim3(32, BL), dim3(256), 0, stream>>>(bufA, attT);

    // 5) adj_att = V att (WMMA)
    k_gemm_vatt<<<dim3(1, 8, BL), dim3(256), 0, stream>>>(Vb, attT, bufB);

    // 6) gauss = exp(-D/2) via cross = x@x (WMMA), reusing bufA
    k_gemm_cross<<<dim3(1, 8, BL), dim3(256), 0, stream>>>(
        xb, xt, rowsq, colsq, bufA);

    // 7) fused double-softmax + gumbel + exact top-k threshold + mask
    k_final<<<dim3(BL), dim3(256), 0, stream>>>(bufB, bufA, gumbel, keys, out);
}
